// AttnVLAD_64269890617691
// MI455X (gfx1250) — matrix operand order repaired and checked by reading, not compiled, not run on hardware
//
#include <hip/hip_runtime.h>
#include <hip/hip_bf16.h>
#include <math.h>

// ---------------------------------------------------------------------------
// AttnVLAD for MI455X (gfx1250, wave32, WMMA).
// Fused: score-GEMM (bf16 WMMA, fp32 acc) -> softmax(K) -> desc-GEMM (bf16
// WMMA) on the same LDS-resident x tile; row-sum renorm is applied after the
// fact as descU/denom (it commutes with the n-sum). x (268MB) is read from
// HBM exactly once -> memory-roofline ~12us; bf16 WMMA keeps compute under
// that.
// ---------------------------------------------------------------------------

typedef __bf16 bf16;
typedef __attribute__((ext_vector_type(16))) __bf16 v16bf;
typedef __attribute__((ext_vector_type(8)))  __bf16 v8bf;
typedef __attribute__((ext_vector_type(4)))  __bf16 v4bf;
typedef __attribute__((ext_vector_type(8)))  float  v8f;

#define B_   16
#define D_   512
#define K_   64
#define N_   8192
#define NT   64          // columns per block iteration
#define NPART 16         // blocks per batch (each handles N/NPART columns)
#define NCOLS (N_ / NPART)

// dynamic-LDS offsets (bytes)
#define OFF_QS   0                    // bf16 [K_][D_]   = 65536
#define OFF_XT   65536                // bf16 [NT][D_]   = 65536  (xT[n][d])
#define OFF_XD   131072               // bf16 [D_][NT]   = 65536  (xd[d][n])
#define OFF_SLS  196608               // f32  [K_][NT]   = 16384
#define OFF_PLS  212992               // bf16 [K_][NT]   = 8192
#define OFF_DACC 221184               // f32  [K_]       = 256
#define LDS_BYTES 221440

union U16 { v16bf v; v8bf h[2]; };

__device__ __forceinline__ v16bf ld_pair(const bf16* c0, const bf16* c1) {
  U16 u;
  u.h[0] = *(const v8bf*)c0;   // ds_load_b128
  u.h[1] = *(const v8bf*)c1;   // ds_load_b128
  return u.v;
}

// ---------------------------------------------------------------------------
__global__ void vlad_zero(float* __restrict__ p, int n) {
  int i = blockIdx.x * blockDim.x + threadIdx.x;
  const int stride = gridDim.x * blockDim.x;
  for (; i < n; i += stride) p[i] = 0.f;
}

// q = l2norm(centers, axis=D), stored transposed+bf16 as qg[k][d]
__global__ __launch_bounds__(256) void vlad_qnorm(const float* __restrict__ centers,
                                                  bf16* __restrict__ qg) {
  __shared__ float red[256];
  const int k = blockIdx.x;
  const int t = threadIdx.x;
  float s = 0.f;
  for (int d = t; d < D_; d += 256) { float v = centers[d * K_ + k]; s += v * v; }
  red[t] = s;
  __syncthreads();
  for (int off = 128; off > 0; off >>= 1) {
    if (t < off) red[t] += red[t + off];
    __syncthreads();
  }
  const float inv = 1.f / fmaxf(sqrtf(red[0]), 1e-12f);
  for (int d = t; d < D_; d += 256) qg[k * D_ + d] = (bf16)(centers[d * K_ + k] * inv);
}

// ---------------------------------------------------------------------------
// Core: per block = (batch b, n-chunk of 512 cols), 8 waves.
//   score  S[k][n]      : A=qs[k][d], B=xT[n][d]   (contract over d)
//   desc   descU^T[k][d]: A=pls[k][n], B=xd[d][n]  (contract over n)
__global__ __launch_bounds__(256, 1) void vlad_core(
    const float* __restrict__ x, const bf16* __restrict__ qg,
    const float* __restrict__ alpha_p,
    float* __restrict__ denom_g, float* __restrict__ descU_g) {
  extern __shared__ char smem[];
  bf16*  qs   = (bf16*)(smem + OFF_QS);
  bf16*  xT   = (bf16*)(smem + OFF_XT);
  bf16*  xd   = (bf16*)(smem + OFF_XD);
  float* sls  = (float*)(smem + OFF_SLS);
  bf16*  pls  = (bf16*)(smem + OFF_PLS);
  float* dacc = (float*)(smem + OFF_DACC);

  const int t    = threadIdx.x;
  const int lane = t & 31;
  const int wave = t >> 5;
  const int b    = blockIdx.x / NPART;
  const int part = blockIdx.x % NPART;
  const int n0b  = part * NCOLS;
  const float alphav = alpha_p[0];

  // stage q (64KB) into LDS
  {
    const uint4* src = (const uint4*)qg;
    uint4* dst = (uint4*)qs;
#pragma unroll
    for (int i = 0; i < 16; ++i) dst[t + 256 * i] = src[t + 256 * i];
  }
  if (t < K_) dacc[t] = 0.f;

  // wave roles
  const int nt_s = wave & 3;   // score n-tile
  const int kp_s = wave >> 2;  // score k-pair -> ktiles {2kp, 2kp+1}
  const int kt_d = wave & 3;   // desc k-tile
  const int dh_d = wave >> 2;  // desc d-half (256 cols of d)

  const v8f vzero = {0.f, 0.f, 0.f, 0.f, 0.f, 0.f, 0.f, 0.f};
  v8f dC[16];
#pragma unroll
  for (int i = 0; i < 16; ++i) dC[i] = vzero;

  const size_t xbase = (size_t)b * D_ * N_;

  for (int it = 0; it < NCOLS / NT; ++it) {
    const int n0 = n0b + it * NT;
    __syncthreads();  // previous iteration's LDS consumers are done

    // ---- load x tile fp32 -> bf16, store both orientations ----
    {
      const int ng = t & 15;   // group of 4 columns
      const int dr = t >> 4;   // 0..15
      if (it + 1 < NCOLS / NT)  // warm next tile (global_prefetch)
        __builtin_prefetch(x + xbase + (size_t)dr * N_ + (n0 + NT), 0, 1);
#pragma unroll 4
      for (int i = 0; i < 32; ++i) {
        const int d = dr + (i << 4);
        const float4 f = *(const float4*)(x + xbase + (size_t)d * N_ + (n0 + 4 * ng));
        const bf16 b0 = (bf16)f.x, b1 = (bf16)f.y, b2 = (bf16)f.z, b3 = (bf16)f.w;
        *(v4bf*)(xd + d * NT + 4 * ng) = (v4bf){b0, b1, b2, b3};
        xT[(4 * ng + 0) * D_ + d] = b0;
        xT[(4 * ng + 1) * D_ + d] = b1;
        xT[(4 * ng + 2) * D_ + d] = b2;
        xT[(4 * ng + 3) * D_ + d] = b3;
      }
    }
    __syncthreads();

    // ---- score GEMM: this wave computes 2 ktiles x 1 ntile ----
    {
      v8f sC0 = vzero, sC1 = vzero;
      const int nrow  = nt_s * 16 + (lane & 15);
      const int dOffB = (lane >> 4) << 4;  // B: lanes>=16 hold K=16..31
      const int dOffA = (lane >> 4) << 4;  // A: lanes>=16 hold K=16..31
      const int k0 = (2 * kp_s + 0) * 16 + (lane & 15);
      const int k1 = (2 * kp_s + 1) * 16 + (lane & 15);
      for (int d0 = 0; d0 < D_; d0 += 32) {
        const bf16* bp = xT + nrow * D_ + d0 + dOffB;
        const v16bf Bv = ld_pair(bp, bp + 8);
        const bf16* a0 = qs + k0 * D_ + d0 + dOffA;
        const v16bf A0 = ld_pair(a0, a0 + 8);
        sC0 = __builtin_amdgcn_wmma_f32_16x16x32_bf16(false, A0, false, Bv,
                                                      (short)0, sC0, false, false);
        const bf16* a1 = qs + k1 * D_ + d0 + dOffA;
        const v16bf A1 = ld_pair(a1, a1 + 8);
        sC1 = __builtin_amdgcn_wmma_f32_16x16x32_bf16(false, A1, false, Bv,
                                                      (short)0, sC1, false, false);
      }
      const int col   = nt_s * 16 + (lane & 15);
      const int rbase = 8 * (lane >> 4);
#pragma unroll
      for (int r = 0; r < 8; ++r) {
        sls[((2 * kp_s + 0) * 16 + rbase + r) * NT + col] = sC0[r];
        sls[((2 * kp_s + 1) * 16 + rbase + r) * NT + col] = sC1[r];
      }
    }
    __syncthreads();

    // ---- softmax over K=64 per column ----
    if (t < NT) {
      const int c = t;
      float m = -3.4e38f;
      for (int k = 0; k < K_; ++k) m = fmaxf(m, alphav * sls[k * NT + c]);
      float s = 0.f;
      for (int k = 0; k < K_; ++k) {
        const float e = __expf(alphav * sls[k * NT + c] - m);
        sls[k * NT + c] = e;
        s += e;
      }
      const float inv = 1.f / s;
      for (int k = 0; k < K_; ++k)
        pls[k * NT + c] = (bf16)(sls[k * NT + c] * inv);
    }
    __syncthreads();

    // denom row-sums over the exact bf16 p fed to the GEMM
    if (t < K_) {
      float rs = 0.f;
      for (int c = 0; c < NT; ++c) rs += (float)pls[t * NT + c];
      dacc[t] += rs;
    }

    // ---- desc GEMM: descU^T[k][d] += p[k][n] * x[d][n] ----
    {
      const int krow  = kt_d * 16 + (lane & 15);
      const int nOffA = (lane >> 4) << 4;
      const int nOffB = (lane >> 4) << 4;
      const int dcol  = dh_d * 256 + (lane & 15);
#pragma unroll
      for (int ns = 0; ns < 2; ++ns) {
        const int na = ns * 32;
        const bf16* ap = pls + krow * NT + na + nOffA;
        const v16bf Av = ld_pair(ap, ap + 8);
#pragma unroll
        for (int dt = 0; dt < 16; ++dt) {
          const bf16* bp = xd + (size_t)(dcol + dt * 16) * NT + na + nOffB;
          const v16bf Bv = ld_pair(bp, bp + 8);
          dC[dt] = __builtin_amdgcn_wmma_f32_16x16x32_bf16(false, Av, false, Bv,
                                                           (short)0, dC[dt], false, false);
        }
      }
    }
  }

  __syncthreads();
  if (t < K_) unsafeAtomicAdd(&denom_g[b * K_ + t], dacc[t]);

  {
    const int rbase = 8 * (lane >> 4);
#pragma unroll
    for (int dt = 0; dt < 16; ++dt) {
      const int dcol = dh_d * 256 + dt * 16 + (lane & 15);
#pragma unroll
      for (int r = 0; r < 8; ++r) {
        const int k = kt_d * 16 + rbase + r;
        unsafeAtomicAdd(&descU_g[((size_t)b * K_ + k) * D_ + dcol], dC[dt][r]);
      }
    }
  }
}

// ---------------------------------------------------------------------------
// finalize: desc = descU/denom - centers; intra-norm over D; *cw; global L2.
__global__ __launch_bounds__(256) void vlad_final(
    const float* __restrict__ descU, const float* __restrict__ denom,
    const float* __restrict__ centers, const float* __restrict__ cw,
    float* __restrict__ out) {
  __shared__ float red[256];
  __shared__ float nk[K_];
  __shared__ float gnorm;
  const int b = blockIdx.x;
  const int t = threadIdx.x;
  const int k = t & 63;
  const int part = t >> 6;  // 0..3, each covers 128 d
  const float dn  = fmaxf(denom[b * K_ + k], 1e-6f);
  const float cwk = cw[k];
  float ss = 0.f;
  for (int d = part * 128; d < part * 128 + 128; ++d) {
    const float v = descU[((size_t)b * K_ + k) * D_ + d] / dn - centers[d * K_ + k];
    ss += v * v;
  }
  red[t] = ss;
  __syncthreads();
  if (t < K_) {
    const float s = red[t] + red[t + 64] + red[t + 128] + red[t + 192];
    const float n = fmaxf(sqrtf(s), 1e-12f);
    nk[t] = n;
    red[t] = cwk * cwk * s / (n * n);
  }
  __syncthreads();
  if (t == 0) {
    float g = 0.f;
    for (int i = 0; i < K_; ++i) g += red[i];
    gnorm = fmaxf(sqrtf(g), 1e-12f);
  }
  __syncthreads();
  const float scale = cwk / (nk[k] * gnorm);
  for (int d = part * 128; d < part * 128 + 128; ++d) {
    const float v = descU[((size_t)b * K_ + k) * D_ + d] / dn - centers[d * K_ + k];
    out[(size_t)b * (D_ * K_) + (size_t)d * K_ + k] = v * scale;
  }
}

// ---------------------------------------------------------------------------
extern "C" void kernel_launch(void* const* d_in, const int* in_sizes, int n_in,
                              void* d_out, int out_size, void* d_ws, size_t ws_size,
                              hipStream_t stream) {
  const float* x       = (const float*)d_in[0];  // [16,512,8192]
  const float* centers = (const float*)d_in[1];  // [1,512,64]
  const float* alpha   = (const float*)d_in[2];  // [1]
  const float* cw      = (const float*)d_in[3];  // [1,1,64]
  float* out = (float*)d_out;                    // [16, 32768]

  char* ws = (char*)d_ws;
  bf16*  qg    = (bf16*)ws;                 // bf16 [64][512]  : 65536 B
  float* denom = (float*)(ws + 65536);      // f32  [16][64]   : 4096 B
  float* descU = (float*)(ws + 69632);      // f32  [16][64][512] : 2 MB

  vlad_zero<<<512, 256, 0, stream>>>(denom, B_ * K_ + B_ * K_ * D_);
  vlad_qnorm<<<K_, 256, 0, stream>>>(centers, qg);
  vlad_core<<<B_ * NPART, 256, LDS_BYTES, stream>>>(x, qg, alpha, denom, descU);
  vlad_final<<<B_, 256, 0, stream>>>(descU, denom, centers, cw, out);
}
